// LigerFusedLinearCrossEntropyHelion_39891656245483
// MI455X (gfx1250) — compile-verified, matrix operand order checked by reading, not run on hardware
//
#include <hip/hip_runtime.h>
#include <hip/hip_bf16.h>

// ---------------- problem constants ----------------
#define BT      4096      // rows of x
#define H       4096      // hidden dim (K)
#define V       32000     // vocab (N)
#define IGNORE  (-100)

// ---------------- tiling ----------------
#define BM      64        // rows per block
#define BN      128       // cols per block iteration
#define KC      128       // K chunk staged in LDS
#define NLOOP   5         // col-block iterations per block (BN*NLOOP = 640 cols)
#define NSPLIT  50        // V / (BN*NLOOP)
#define KCHUNKS (H / KC)  // 32
#define NCHUNK  (NLOOP * KCHUNKS)
#define LDS_PAD 8         // bf16 elements of row padding (16B) -> kills bank conflicts

typedef __bf16 bf16_t;
typedef __bf16 bf16x8  __attribute__((ext_vector_type(8)));
typedef __bf16 bf16x16 __attribute__((ext_vector_type(16)));
typedef float  v8f     __attribute__((ext_vector_type(8)));

union FragU { bf16x16 v; bf16x8 h[2]; };

static __device__ __forceinline__ bf16x8 cvt8(float4 a, float4 b) {
    bf16x8 r;
    r[0] = (bf16_t)a.x; r[1] = (bf16_t)a.y; r[2] = (bf16_t)a.z; r[3] = (bf16_t)a.w;
    r[4] = (bf16_t)b.x; r[5] = (bf16_t)b.y; r[6] = (bf16_t)b.z; r[7] = (bf16_t)b.w;
    return r;
}

// =====================================================================
// Kernel 1: bf16 WMMA GEMM tile + online log-sum-exp partials.
// grid = (BT/BM, NSPLIT), block = 256 threads (8 waves of 32).
// Wave (wr = wid>>1, wc = wid&1): rows [wr*16, wr*16+16), cols wc*64 + 4 tiles.
// Double-buffered LDS (105 KB of the 320 KB WGP pool): next K-chunk's global
// loads are issued before the current chunk's 16 WMMAs, converted+stored
// after, with ONE barrier per chunk.
// =====================================================================
__global__ void __launch_bounds__(256)
flce_gemm_lse_partial(const float* __restrict__ x,
                      const float* __restrict__ w,
                      float2* __restrict__ partials) {
    __shared__ __align__(16) bf16_t xs [2][BM][KC + LDS_PAD];
    __shared__ __align__(16) bf16_t wsh[2][BN][KC + LDS_PAD];
    __shared__ float stats[2][4][16][2];   // [wc][wr][row-in-tile][{m,s}]

    const int tid  = threadIdx.x;
    const int lane = tid & 31;
    const int wid  = tid >> 5;
    const int wr   = wid >> 1;          // 0..3 row-tile of this wave
    const int wc   = wid &  1;          // 0..1 col-half of this wave
    const int lr   = lane & 15;         // lane within 16-group
    const int h8   = (lane >> 4) * 8;   // K sub-offset per wave half

    const int row0     = blockIdx.x * BM;
    const int col_base = blockIdx.y * (BN * NLOOP);

    // staging coordinates (32 x-elems, 64 w-elems per thread)
    const int xr  = tid >> 2;
    const int xc  = (tid & 3) * 32;
    const int wwr = tid >> 1;
    const int wwc = (tid & 1) * 64;

    // running per-row online-softmax state (row = wr*16 + j + (lane>=16)*8)
    float rm[8], rs[8];
#pragma unroll
    for (int j = 0; j < 8; ++j) { rm[j] = -__builtin_inff(); rs[j] = 0.0f; }

    v8f acc[4];
#pragma unroll
    for (int t = 0; t < 4; ++t) { v8f z = {}; acc[t] = z; }

    float4 xreg[8];
    float4 wreg[16];

    // ---- prologue: stage chunk 0 into buffer 0 ----
    {
        const float4* px = (const float4*)(x + (size_t)(row0 + xr) * H + xc);
        const float4* pw = (const float4*)(w + (size_t)(col_base + wwr) * H + wwc);
#pragma unroll
        for (int u = 0; u < 8; ++u)  xreg[u] = px[u];
#pragma unroll
        for (int u = 0; u < 16; ++u) wreg[u] = pw[u];
        bf16x8* sx = (bf16x8*)&xs[0][xr][xc];
#pragma unroll
        for (int u = 0; u < 4; ++u) sx[u] = cvt8(xreg[2 * u], xreg[2 * u + 1]);
        bf16x8* sw = (bf16x8*)&wsh[0][wwr][wwc];
#pragma unroll
        for (int u = 0; u < 8; ++u) sw[u] = cvt8(wreg[2 * u], wreg[2 * u + 1]);
    }
    __syncthreads();

    int cur = 0;
    for (int it = 0; it < NCHUNK; ++it) {
        const bool has_next = (it + 1) < NCHUNK;

        // ---- issue next chunk's global loads (latency hidden by WMMAs) ----
        if (has_next) {
            const int nb2 = (it + 1) >> 5;               // KCHUNKS == 32
            const int k02 = ((it + 1) & 31) * KC;
            const int col2 = col_base + nb2 * BN;
            const float4* px = (const float4*)(x + (size_t)(row0 + xr) * H + k02 + xc);
            const float4* pw = (const float4*)(w + (size_t)(col2 + wwr) * H + k02 + wwc);
#pragma unroll
            for (int u = 0; u < 8; ++u)  xreg[u] = px[u];
#pragma unroll
            for (int u = 0; u < 16; ++u) wreg[u] = pw[u];
        }

        // ---- 16 WMMAs on the current buffer ----
#pragma unroll
        for (int kk = 0; kk < KC; kk += 32) {
            FragU A;
            const int arow = wr * 16 + lr;
            A.h[0] = *(const bf16x8*)&xs[cur][arow][kk + h8];
            A.h[1] = *(const bf16x8*)&xs[cur][arow][kk + 16 + h8];
            FragU B0, B1, B2, B3;
            const int brow = wc * 64 + lr;
            B0.h[0] = *(const bf16x8*)&wsh[cur][brow     ][kk + h8];
            B0.h[1] = *(const bf16x8*)&wsh[cur][brow     ][kk + 16 + h8];
            B1.h[0] = *(const bf16x8*)&wsh[cur][brow + 16][kk + h8];
            B1.h[1] = *(const bf16x8*)&wsh[cur][brow + 16][kk + 16 + h8];
            B2.h[0] = *(const bf16x8*)&wsh[cur][brow + 32][kk + h8];
            B2.h[1] = *(const bf16x8*)&wsh[cur][brow + 32][kk + 16 + h8];
            B3.h[0] = *(const bf16x8*)&wsh[cur][brow + 48][kk + h8];
            B3.h[1] = *(const bf16x8*)&wsh[cur][brow + 48][kk + 16 + h8];
            acc[0] = __builtin_amdgcn_wmma_f32_16x16x32_bf16(
                false, A.v, false, B0.v, (short)0, acc[0], false, false);
            acc[1] = __builtin_amdgcn_wmma_f32_16x16x32_bf16(
                false, A.v, false, B1.v, (short)0, acc[1], false, false);
            acc[2] = __builtin_amdgcn_wmma_f32_16x16x32_bf16(
                false, A.v, false, B2.v, (short)0, acc[2], false, false);
            acc[3] = __builtin_amdgcn_wmma_f32_16x16x32_bf16(
                false, A.v, false, B3.v, (short)0, acc[3], false, false);
        }

        // ---- end of a col-block: per-row (m,s) over 64 cols, online merge ----
        if ((it & 31) == 31) {
#pragma unroll
            for (int j = 0; j < 8; ++j) {
                float mt = -__builtin_inff();
#pragma unroll
                for (int t = 0; t < 4; ++t) mt = fmaxf(mt, acc[t][j]);
#pragma unroll
                for (int m = 8; m >= 1; m >>= 1)
                    mt = fmaxf(mt, __shfl_xor(mt, m, 32));
                float st = 0.0f;
#pragma unroll
                for (int t = 0; t < 4; ++t) st += __expf(acc[t][j] - mt);
#pragma unroll
                for (int m = 8; m >= 1; m >>= 1)
                    st += __shfl_xor(st, m, 32);
                const float nm = fmaxf(rm[j], mt);
                rs[j] = rs[j] * __expf(rm[j] - nm) + st * __expf(mt - nm);
                rm[j] = nm;
            }
#pragma unroll
            for (int t = 0; t < 4; ++t) { v8f z = {}; acc[t] = z; }
        }

        // ---- convert + store next chunk into the other buffer ----
        if (has_next) {
            bf16x8* sx = (bf16x8*)&xs[cur ^ 1][xr][xc];
#pragma unroll
            for (int u = 0; u < 4; ++u) sx[u] = cvt8(xreg[2 * u], xreg[2 * u + 1]);
            bf16x8* sw = (bf16x8*)&wsh[cur ^ 1][wwr][wwc];
#pragma unroll
            for (int u = 0; u < 8; ++u) sw[u] = cvt8(wreg[2 * u], wreg[2 * u + 1]);
        }
        __syncthreads();
        cur ^= 1;
    }

    // ---- cross-wave combine + write partials ----
    if (lr == 0) {
        const int hb = (lane >> 4) * 8;
#pragma unroll
        for (int j = 0; j < 8; ++j) {
            stats[wc][wr][j + hb][0] = rm[j];
            stats[wc][wr][j + hb][1] = rs[j];
        }
    }
    __syncthreads();
    if (tid < BM) {
        const int r  = tid;
        const int tw = r >> 4, tr = r & 15;
        const float m0 = stats[0][tw][tr][0], s0 = stats[0][tw][tr][1];
        const float m1 = stats[1][tw][tr][0], s1 = stats[1][tw][tr][1];
        const float m  = fmaxf(m0, m1);
        const float s  = s0 * __expf(m0 - m) + s1 * __expf(m1 - m);
        float2 p; p.x = m; p.y = s;
        partials[(size_t)(row0 + r) * NSPLIT + blockIdx.y] = p;
    }
}

// =====================================================================
// Kernel 2: exact fp32 target logit: tgt[r] = dot(x[r], weight[clip(t[r])]).
// One wave per row; grid = BT/8, block = 256.
// =====================================================================
__global__ void __launch_bounds__(256)
flce_target_logit(const float* __restrict__ x,
                  const float* __restrict__ w,
                  const long long* __restrict__ target,
                  float* __restrict__ tgt_out) {
    const int lane = threadIdx.x & 31;
    const int wid  = threadIdx.x >> 5;
    const int row  = blockIdx.x * 8 + wid;
    if (row >= BT) return;

    long long t = target[row];
    t = (t < 0) ? 0 : (t >= V ? (V - 1) : t);

    const float4* px = (const float4*)(x + (size_t)row * H);
    const float4* pw = (const float4*)(w + (size_t)t * H);

    float s = 0.0f;
#pragma unroll 4
    for (int i = 0; i < H / 128; ++i) {
        const int idx = i * 32 + lane;     // float4 index
        float4 a = px[idx];
        float4 b = pw[idx];
        s = fmaf(a.x, b.x, s);
        s = fmaf(a.y, b.y, s);
        s = fmaf(a.z, b.z, s);
        s = fmaf(a.w, b.w, s);
    }
#pragma unroll
    for (int m = 16; m >= 1; m >>= 1) s += __shfl_xor(s, m, 32);
    if (lane == 0) tgt_out[row] = s;
}

// =====================================================================
// Kernel 3: combine partials -> lse, nll, valid count, final loss.
// Single block of 1024 threads.
// =====================================================================
__global__ void __launch_bounds__(1024)
flce_finalize(const long long* __restrict__ target,
              const float2* __restrict__ partials,
              const float* __restrict__ tgt,
              float* __restrict__ out) {
    __shared__ float sf[1024];
    __shared__ int   si[1024];

    const int tid = threadIdx.x;
    float nll = 0.0f;
    int   cnt = 0;

    for (int r = tid; r < BT; r += 1024) {
        float m = -__builtin_inff();
        const float2* p = partials + (size_t)r * NSPLIT;
#pragma unroll 2
        for (int s = 0; s < NSPLIT; ++s) m = fmaxf(m, p[s].x);
        float ssum = 0.0f;
#pragma unroll 2
        for (int s = 0; s < NSPLIT; ++s) ssum += p[s].y * __expf(p[s].x - m);
        const float lse = m + __logf(ssum);
        if (target[r] != IGNORE) {
            nll += lse - tgt[r];
            cnt += 1;
        }
    }
    sf[tid] = nll;
    si[tid] = cnt;
    __syncthreads();
    for (int off = 512; off > 0; off >>= 1) {
        if (tid < off) { sf[tid] += sf[tid + off]; si[tid] += si[tid + off]; }
        __syncthreads();
    }
    if (tid == 0) {
        const float n = (float)si[0];
        out[0] = (n > 0.0f) ? (sf[0] / n) : 0.0f;
    }
}

// =====================================================================
extern "C" void kernel_launch(void* const* d_in, const int* in_sizes, int n_in,
                              void* d_out, int out_size, void* d_ws, size_t ws_size,
                              hipStream_t stream) {
    const float*     x      = (const float*)d_in[0];
    const float*     weight = (const float*)d_in[1];
    const long long* target = (const long long*)d_in[2];
    float*           out    = (float*)d_out;

    float2* partials = (float2*)d_ws;                                   // BT*NSPLIT float2
    float*  tgtlog   = (float*)((char*)d_ws +
                                (size_t)BT * NSPLIT * sizeof(float2));  // BT floats

    dim3 g1(BT / BM, NSPLIT);
    flce_gemm_lse_partial<<<g1, 256, 0, stream>>>(x, weight, partials);

    flce_target_logit<<<BT / 8, 256, 0, stream>>>(x, weight, target, tgtlog);

    flce_finalize<<<1, 1024, 0, stream>>>(target, partials, tgtlog, out);
}
